// DeepFM_58016418234670
// MI455X (gfx1250) — compile-verified, matrix operand order checked by reading, not run on hardware
//
#include <hip/hip_runtime.h>
#include <hip/hip_bf16.h>

typedef __attribute__((ext_vector_type(16))) _Float16 v16h;
typedef __attribute__((ext_vector_type(8)))  _Float16 v8h;
typedef __attribute__((ext_vector_type(4)))  _Float16 v4h;
typedef __attribute__((ext_vector_type(8)))  float    v8f;

#define B_ROWS   131072
#define NFEAT    17
#define K1       1088
#define MT       64          // rows per block
#define FSTRIDE  1096        // padded f16 stride of feats rows
#define H_STRIDE 264         // padded f16 stride for h1/h2 (256 cols)
#define H3_STRIDE 136        // padded f16 stride for h3 (128 cols)

// cumsum of VOCABS (feature order: 8 numeric, then categorical cols 16..8)
__constant__ int c_offsets[17] = {0,64,80,208,272,400,464,976,1488,
                                  15089,15100,29404,63247,66392,79562,92635,98078};

__device__ __forceinline__ int feat_col(int f) { return f < 8 ? f : 24 - f; }

// Build a 16x32 f16 A/B fragment per CDNA5 layout:
// lane l holds row (l&15); K = kb..kb+7 in regs 0-3, K = kb+16..kb+23 in regs 4-7,
// where kb = k0 + (l>>4)*8.  Two contiguous 16-byte loads.
__device__ __forceinline__ v16h load_frag(const _Float16* row, int kb) {
  v8h lo = *(const v8h*)(row + kb);
  v8h hi = *(const v8h*)(row + kb + 16);
  v16h r;
#pragma unroll
  for (int i = 0; i < 8; ++i) { r[i] = lo[i]; r[8 + i] = hi[i]; }
  return r;
}

// One MLP layer tile-set for a wave: NT n-tiles (16 cols each) x 4 m-tiles (64 rows),
// A from LDS (row-major f16), Bt from global ([N][K] f16), fused bias + leaky-relu,
// f16 result to LDS.
template <int NT>
__device__ __forceinline__ void gemm_stage(const _Float16* __restrict__ A, int astride,
                                           const _Float16* __restrict__ Bt, int K,
                                           const float* __restrict__ bias,
                                           _Float16* __restrict__ outp, int ostride,
                                           int lane, int nt0) {
  v8f acc[NT][4] = {};
  const int lm = lane & 15;
  const int lk = (lane >> 4) * 8;
  for (int k0 = 0; k0 < K; k0 += 32) {
    v16h bf[NT];
#pragma unroll
    for (int j = 0; j < NT; ++j)
      bf[j] = load_frag(Bt + (size_t)((nt0 + j) * 16 + lm) * K, k0 + lk);
    v16h af[4];
#pragma unroll
    for (int mt = 0; mt < 4; ++mt)
      af[mt] = load_frag(A + (mt * 16 + lm) * astride, k0 + lk);
#pragma unroll
    for (int j = 0; j < NT; ++j)
#pragma unroll
      for (int mt = 0; mt < 4; ++mt)
        acc[j][mt] = __builtin_amdgcn_wmma_f32_16x16x32_f16(
            false, af[mt], false, bf[j], (short)0, acc[j][mt], false, false);
  }
  // C/D layout: VGPR v, lanes 0-15 -> M=v, lanes 16-31 -> M=v+8; N = lane&15
#pragma unroll
  for (int j = 0; j < NT; ++j) {
    float bn = bias[(nt0 + j) * 16 + lm];
#pragma unroll
    for (int mt = 0; mt < 4; ++mt)
#pragma unroll
      for (int v = 0; v < 8; ++v) {
        float x = acc[j][mt][v] + bn;
        x = x > 0.f ? x : 0.01f * x;  // leaky relu
        outp[(mt * 16 + (lane >> 4) * 8 + v) * ostride + (nt0 + j) * 16 + lm] = (_Float16)x;
      }
  }
}

// Convert + transpose weights to f16 [N][K] row-major in workspace.
__global__ void prep_weights(const float* __restrict__ w1, const float* __restrict__ w2,
                             const float* __restrict__ w3,
                             _Float16* __restrict__ w1t, _Float16* __restrict__ w2t,
                             _Float16* __restrict__ w3t) {
  int i = blockIdx.x * blockDim.x + threadIdx.x;
  const int E1 = 256 * 1088, E2 = 256 * 256, E3 = 128 * 256;
  if (i < E1) {
    int n = i / 1088, k = i % 1088;
    w1t[i] = (_Float16)w1[k * 256 + n];
  } else if (i < E1 + E2) {
    int j = i - E1, n = j / 256, k = j % 256;
    w2t[j] = (_Float16)w2[k * 256 + n];
  } else if (i < E1 + E2 + E3) {
    int j = i - E1 - E2, n = j / 256, k = j % 256;
    w3t[j] = (_Float16)w3[k * 128 + n];
  }
}

__global__ __launch_bounds__(256) void deepfm_fused(
    const int* __restrict__ x, const float* __restrict__ table,
    const float* __restrict__ bias_table,
    const _Float16* __restrict__ w1t, const float* __restrict__ b1,
    const _Float16* __restrict__ w2t, const float* __restrict__ b2,
    const _Float16* __restrict__ w3t, const float* __restrict__ b3,
    const float* __restrict__ w4, const float* __restrict__ b4,
    float* __restrict__ out) {
  extern __shared__ char smem[];
  _Float16* feats  = (_Float16*)smem;               // [64][1096] f16 (140288 B)
  _Float16* h1     = (_Float16*)(smem + 140288);    // [64][264]  f16 (33792 B)
  _Float16* h2     = (_Float16*)(smem + 174080);    // [64][264]  f16 (33792 B)
  _Float16* h3     = (_Float16*)smem;               // [64][136]  f16 (reuses feats)
  float*    fm_lds = (float*)(smem + 207872);       // [64]
  float*    bi_lds = (float*)(smem + 208128);       // [64]

  const int t = threadIdx.x;
  const int R = blockIdx.x * MT;

  // ---- Phase 0a: gather 17 embeddings/row -> f16 LDS (A matrix for GEMM1)
  {
    const int g = t >> 4, q = t & 15;  // 16 groups of 16 lanes; 4 floats/lane
    for (int p = g; p < MT * NFEAT; p += 16) {
      int r = p / NFEAT, f = p % NFEAT;
      long idx = (long)x[(R + r) * 18 + feat_col(f)] + c_offsets[f];
      const float4 v = *(const float4*)(table + idx * 64 + q * 4);
      v4h hv = {(_Float16)v.x, (_Float16)v.y, (_Float16)v.z, (_Float16)v.w};
      *(v4h*)(feats + r * FSTRIDE + f * 64 + q * 4) = hv;
    }
  }
  __syncthreads();

  // ---- Phase 0b: FM second order + linear bias term
  {
    const int r = t >> 2, d0 = (t & 3) * 16;
    float partial = 0.f;
#pragma unroll
    for (int dd = 0; dd < 16; ++dd) {
      int d = d0 + dd;
      float s = 0.f, sq = 0.f;
      for (int f = 0; f < NFEAT; ++f) {
        float v = (float)feats[r * FSTRIDE + f * 64 + d];
        s += v; sq += v * v;
      }
      partial += 0.5f * (s * s - sq);
    }
    partial += __shfl_xor(partial, 1);
    partial += __shfl_xor(partial, 2);
    if ((t & 3) == 0) fm_lds[r] = partial;
    if (t < MT) {
      float bs = 0.f;
      for (int f = 0; f < NFEAT; ++f) {
        long idx = (long)x[(R + t) * 18 + feat_col(f)] + c_offsets[f];
        bs += bias_table[idx];
      }
      bi_lds[t] = bs;
    }
  }
  __syncthreads();

  const int wave = t >> 5, lane = t & 31;
  // ---- GEMM1: [64x1088] x [1088x256], wave owns 2 n-tiles x 4 m-tiles
  gemm_stage<2>(feats, FSTRIDE, w1t, K1, b1, h1, H_STRIDE, lane, wave * 2);
  __syncthreads();
  // ---- GEMM2: [64x256] x [256x256]
  gemm_stage<2>(h1, H_STRIDE, w2t, 256, b2, h2, H_STRIDE, lane, wave * 2);
  __syncthreads();
  // ---- GEMM3: [64x256] x [256x128], wave owns 1 n-tile
  gemm_stage<1>(h2, H_STRIDE, w3t, 256, b3, h3, H3_STRIDE, lane, wave);
  __syncthreads();

  // ---- Final 128 -> 1 + combine (4 lanes/row, shuffle reduce)
  {
    const int r = t >> 2, q = t & 3;
    float dot = 0.f;
#pragma unroll
    for (int i = 0; i < 32; ++i)
      dot += (float)h3[r * H3_STRIDE + q * 32 + i] * w4[q * 32 + i];
    dot += __shfl_xor(dot, 1);
    dot += __shfl_xor(dot, 2);
    if (q == 0) out[R + r] = dot + b4[0] + bi_lds[r] + fm_lds[r];
  }
}

extern "C" void kernel_launch(void* const* d_in, const int* in_sizes, int n_in,
                              void* d_out, int out_size, void* d_ws, size_t ws_size,
                              hipStream_t stream) {
  const int*   x          = (const int*)d_in[0];
  const float* table      = (const float*)d_in[1];
  const float* bias_table = (const float*)d_in[2];
  const float* w1 = (const float*)d_in[3];
  const float* b1 = (const float*)d_in[4];
  const float* w2 = (const float*)d_in[5];
  const float* b2 = (const float*)d_in[6];
  const float* w3 = (const float*)d_in[7];
  const float* b3 = (const float*)d_in[8];
  const float* w4 = (const float*)d_in[9];
  const float* b4 = (const float*)d_in[10];
  float* out = (float*)d_out;

  // workspace: f16 transposed weights (w1t 544KB + w2t 128KB + w3t 64KB)
  _Float16* w1t = (_Float16*)d_ws;
  _Float16* w2t = w1t + 256 * 1088;
  _Float16* w3t = w2t + 256 * 256;

  const int total = 256 * 1088 + 256 * 256 + 128 * 256;
  prep_weights<<<(total + 255) / 256, 256, 0, stream>>>(w1, w2, w3, w1t, w2t, w3t);

  const size_t dyn_lds = 208128 + 256;  // feats + h1 + h2 + fm + bias
  deepfm_fused<<<dim3(B_ROWS / MT), 256, dyn_lds, stream>>>(
      x, table, bias_table, w1t, b1, w2t, b2, w3t, b3, w4, b4, out);
}
